// CountryAttnEncoder_23828478558673
// MI455X (gfx1250) — matrix operand order repaired and checked by reading, not compiled, not run on hardware
//
#include <hip/hip_runtime.h>
#include <hip/hip_bf16.h>

typedef __attribute__((ext_vector_type(16))) _Float16 v16h;
typedef __attribute__((ext_vector_type(8)))  float    v8f;

#define S_TOK 86
#define DIM   64
#define NH    4
#define HD    16
#define NREG  7
#define NFEAT 11
#define HIDN  128
#define QKVN  192
#define MT    6          // ceil(86/16)

union Frag { v16h h; unsigned u[8]; _Float16 e[16]; };

// ---- WMMA fragment loaders (wave32, contiguous K order k = 16*half + s, same in A and B) ----

// A matrix 16x32 f16: lane m = lane%16, half=lane/16,
// VGPR i holds K pair starting at 2*i + 16*half
__device__ __forceinline__ v16h load_A(const _Float16* base, int stride, int m0,
                                       int k0, int mvalid, int kvalid) {
  const int lane = threadIdx.x & 31;
  const int half = lane >> 4;
  const int m = m0 + (lane & 15);
  Frag r;
#pragma unroll
  for (int i = 0; i < 8; ++i) {
    const int k = k0 + 2 * i + 16 * half;
    unsigned u = 0u;
    if (m < mvalid && k < kvalid)
      u = *(const unsigned*)(base + m * stride + k);   // K pairs contiguous
    r.u[i] = u;
  }
  return r.h;
}

// B matrix 32x16 f16 from an N-major buffer (element (k,n) at base[n*stride+k]):
// lane n = lane%16, VGPR i holds K pair starting at 2*i + 16*(lane/16)
__device__ __forceinline__ v16h load_Bt(const _Float16* base, int stride, int n0,
                                        int k0, int nvalid, int kvalid) {
  const int lane = threadIdx.x & 31;
  const int half = lane >> 4;
  const int n = n0 + (lane & 15);
  Frag r;
#pragma unroll
  for (int i = 0; i < 8; ++i) {
    const int k = k0 + 2 * i + 16 * half;
    unsigned u = 0u;
    if (n < nvalid && k < kvalid)
      u = *(const unsigned*)(base + n * stride + k);
    r.u[i] = u;
  }
  return r.h;
}

// B matrix 32x16 f16 from a K-major buffer (element (k,n) at base[k*stride+n])
__device__ __forceinline__ v16h load_Bcol(const _Float16* base, int stride, int n0,
                                          int k0, int kvalid) {
  const int lane = threadIdx.x & 31;
  const int half = lane >> 4;
  const int n = n0 + (lane & 15);
  Frag r;
#pragma unroll
  for (int i = 0; i < 8; ++i) {
    const int k = k0 + 2 * i + 16 * half;
    r.e[2 * i]     = (k < kvalid)     ? base[k * stride + n]       : (_Float16)0.f;
    r.e[2 * i + 1] = (k + 1 < kvalid) ? base[(k + 1) * stride + n] : (_Float16)0.f;
  }
  return r.h;
}

// C/D 16x16 f32: VGPR v -> row m = v + 8*(lane/16), col n = lane%16
__device__ __forceinline__ void store_C_f16(v8f c, _Float16* base, int stride,
                                            int m0, int n0, int mvalid) {
  const int lane = threadIdx.x & 31;
  const int half = lane >> 4;
  const int n = n0 + (lane & 15);
#pragma unroll
  for (int v = 0; v < 8; ++v) {
    const int m = m0 + v + 8 * half;
    if (m < mvalid) base[m * stride + n] = (_Float16)c[v];
  }
}

__device__ __forceinline__ v8f wmma16(v16h a, v16h b, v8f c) {
  return __builtin_amdgcn_wmma_f32_16x16x32_f16(false, a, false, b, (short)0, c,
                                                false, false);
}

// ---- prep: transpose + f16-convert shared weights once per launch ----
__global__ void ca_prep_kernel(const float* __restrict__ Wqkv,
                               const float* __restrict__ Wa,
                               _Float16* __restrict__ ws) {
  int idx = blockIdx.x * 256 + threadIdx.x;
  if (idx < DIM * QKVN) {                 // Wqkv (64x192) -> ws[n*64+k]
    int k = idx / QKVN, n = idx % QKVN;
    ws[n * DIM + k] = (_Float16)Wqkv[idx];
  }
  if (idx < DIM * DIM) {                  // Wa (64x64) -> ws[Q + n*64+k]
    int k = idx >> 6, n = idx & 63;
    ws[QKVN * DIM + n * DIM + k] = (_Float16)Wa[idx];
  }
}

// ---- main fused kernel: one block (4 waves) per batch element ----
__global__ void __launch_bounds__(128)
ca_main_kernel(const float* __restrict__ influence,
               const float* __restrict__ cstat,
               const float* __restrict__ Wc,  const float* __restrict__ bc,
               const float* __restrict__ bqkv, const float* __restrict__ ba,
               const float* __restrict__ Wo,  const float* __restrict__ bo,
               const _Float16* __restrict__ wqkv_t,
               const _Float16* __restrict__ wa_t,
               float* __restrict__ out) {
  __shared__ __align__(16) _Float16 sh_qkv[S_TOK * QKVN];   // 33.0 KB
  __shared__ __align__(16) _Float16 sh_tok[S_TOK * DIM];    // 10.75 KB (reused as ctx)
  __shared__ __align__(16) _Float16 sh_attn[NH * 16 * 96];  // 12 KB (per-wave 16x96)
  __shared__ float sh_concat[(1 + NREG) * DIM];
  __shared__ int   sh_region[S_TOK];
  __shared__ float sh_cnt[NREG];

  const int tid  = threadIdx.x;
  const int w    = tid >> 5;
  const int lane = tid & 31;
  const int half = lane >> 4;
  const int b    = blockIdx.x;
  _Float16* sh_ctx = sh_tok;

  // stage 0: region ids
  if (tid < S_TOK) {
    int r = 0;
#pragma unroll
    for (int j = 0; j < NREG; ++j)
      if (cstat[tid * NFEAT + 2 + j] > 0.5f) r = j;
    sh_region[tid] = r;
  }
  __syncthreads();
  if (tid < NREG) {
    float c = 0.f;
    for (int s = 0; s < S_TOK; ++s) c += (sh_region[s] == tid) ? 1.f : 0.f;
    sh_cnt[tid] = fmaxf(c, 1.f);
  }

  // stage 1: tokens = relu(feats @ Wc + bc), scalar (K=13)
  for (int idx = tid; idx < S_TOK * DIM; idx += 128) {
    const int s = idx >> 6, d = idx & 63;
    const float f0 = influence[b * (2 * S_TOK) + s] * 0.1f;
    const float f1 = influence[b * (2 * S_TOK) + S_TOK + s] * 0.1f;
    float acc = bc[d] + f0 * Wc[d] + f1 * Wc[DIM + d];
#pragma unroll
    for (int j = 0; j < NFEAT; ++j)
      acc += cstat[s * NFEAT + j] * Wc[(2 + j) * DIM + d];
    sh_tok[idx] = (_Float16)fmaxf(acc, 0.f);
  }
  __syncthreads();

  // stage 2: qkv = tokens @ Wqkv + bqkv  (M=86, N=192, K=64) via WMMA
  // B fragments + biases hoisted into registers; A loaded once per row-block.
  {
    v16h Bq[3][2];
    float bias[3];
#pragma unroll
    for (int nn = 0; nn < 3; ++nn) {
      const int nt = w * 3 + nn;
      Bq[nn][0] = load_Bt(wqkv_t, DIM, nt * 16, 0, QKVN, DIM);
      Bq[nn][1] = load_Bt(wqkv_t, DIM, nt * 16, 32, QKVN, DIM);
      bias[nn]  = bqkv[nt * 16 + (lane & 15)];
    }
    for (int mt = 0; mt < MT; ++mt) {
      const v16h A0 = load_A(sh_tok, DIM, mt * 16, 0, S_TOK, DIM);
      const v16h A1 = load_A(sh_tok, DIM, mt * 16, 32, S_TOK, DIM);
#pragma unroll
      for (int nn = 0; nn < 3; ++nn) {
        v8f acc = {};
        acc = wmma16(A0, Bq[nn][0], acc);
        acc = wmma16(A1, Bq[nn][1], acc);
#pragma unroll
        for (int v = 0; v < 8; ++v) acc[v] += bias[nn];
        store_C_f16(acc, sh_qkv, QKVN, mt * 16, (w * 3 + nn) * 16, S_TOK);
      }
    }
  }
  __syncthreads();

  // stage 3: attention, one head per wave; K/V fragments hoisted (resident
  // across all 6 query row-blocks)
  {
    const int h = w;
    const _Float16* qb = sh_qkv + 16 * h;
    const _Float16* kb = sh_qkv + DIM + 16 * h;
    const _Float16* vb = sh_qkv + 2 * DIM + 16 * h;
    _Float16* at = sh_attn + w * 16 * 96;

    v16h Bk[MT];
#pragma unroll
    for (int kt = 0; kt < MT; ++kt)
      Bk[kt] = load_Bt(kb, QKVN, kt * 16, 0, S_TOK, HD);
    v16h Bv[3];
#pragma unroll
    for (int kc = 0; kc < 3; ++kc)
      Bv[kc] = load_Bcol(vb, QKVN, 0, kc * 32, S_TOK);

    for (int mt = 0; mt < MT; ++mt) {
      const v16h Aq = load_A(qb, QKVN, mt * 16, 0, S_TOK, HD);
      v8f sc[MT];
#pragma unroll
      for (int kt = 0; kt < MT; ++kt) {
        v8f z = {};
        sc[kt] = wmma16(Aq, Bk[kt], z);
      }
      // softmax over 96 cols (cols >= 86 masked)
      float mx[8], sm[8];
#pragma unroll
      for (int v = 0; v < 8; ++v) mx[v] = -1e30f;
#pragma unroll
      for (int kt = 0; kt < MT; ++kt)
#pragma unroll
        for (int v = 0; v < 8; ++v) {
          float x = sc[kt][v] * 0.25f;                    // 1/sqrt(16)
          if (kt * 16 + (lane & 15) >= S_TOK) x = -1e30f;
          sc[kt][v] = x;
          mx[v] = fmaxf(mx[v], x);
        }
#pragma unroll
      for (int off = 1; off < 16; off <<= 1)
#pragma unroll
        for (int v = 0; v < 8; ++v) mx[v] = fmaxf(mx[v], __shfl_xor(mx[v], off));
#pragma unroll
      for (int v = 0; v < 8; ++v) sm[v] = 0.f;
#pragma unroll
      for (int kt = 0; kt < MT; ++kt)
#pragma unroll
        for (int v = 0; v < 8; ++v) {
          const float e = __expf(sc[kt][v] - mx[v]);
          sc[kt][v] = e;
          sm[v] += e;
        }
#pragma unroll
      for (int off = 1; off < 16; off <<= 1)
#pragma unroll
        for (int v = 0; v < 8; ++v) sm[v] += __shfl_xor(sm[v], off);
      // write 16x96 attn block (f16) to per-wave scratch
#pragma unroll
      for (int kt = 0; kt < MT; ++kt)
#pragma unroll
        for (int v = 0; v < 8; ++v)
          at[(v + 8 * half) * 96 + kt * 16 + (lane & 15)] =
              (_Float16)(sc[kt][v] / sm[v]);
      // ctx tile: attn(16x96) @ v(96x16); same-wave LDS is in-order
      v8f ct = {};
#pragma unroll
      for (int kc = 0; kc < 3; ++kc)
        ct = wmma16(load_A(at, 96, 0, kc * 32, 16, 96), Bv[kc], ct);
      store_C_f16(ct, sh_ctx, DIM, mt * 16, 16 * h, S_TOK);
    }
  }
  __syncthreads();

  // stage 4: attn_out = ctx @ Wa + ba, with fused global/region pooling
  {
    const v16h Bw0 = load_Bt(wa_t, DIM, w * 16, 0, DIM, DIM);
    const v16h Bw1 = load_Bt(wa_t, DIM, w * 16, 32, DIM, DIM);
    const float bias = ba[w * 16 + (lane & 15)];
    float pool[1 + NREG];
#pragma unroll
    for (int g = 0; g <= NREG; ++g) pool[g] = 0.f;
#pragma unroll
    for (int mt = 0; mt < MT; ++mt) {
      v8f acc = {};
      acc = wmma16(load_A(sh_ctx, DIM, mt * 16, 0, S_TOK, DIM), Bw0, acc);
      acc = wmma16(load_A(sh_ctx, DIM, mt * 16, 32, S_TOK, DIM), Bw1, acc);
#pragma unroll
      for (int v = 0; v < 8; ++v) {
        const int m = mt * 16 + v + 8 * half;
        if (m < S_TOK) {
          const float x = acc[v] + bias;
          pool[0] += x;
          const int r = sh_region[m];
#pragma unroll
          for (int j = 0; j < NREG; ++j) pool[1 + j] += (j == r) ? x : 0.f;
        }
      }
    }
#pragma unroll
    for (int g = 0; g <= NREG; ++g) pool[g] += __shfl_xor(pool[g], 16);
    if (lane < 16) {
      const int d = w * 16 + lane;
      sh_concat[d] = pool[0] * (1.f / (float)S_TOK);
#pragma unroll
      for (int j = 0; j < NREG; ++j)
        sh_concat[(1 + j) * DIM + d] = pool[1 + j] / sh_cnt[j];
    }
  }
  __syncthreads();

  // stage 5: out = relu(concat(512) @ Wo + bo); thread per output column
  {
    float acc = bo[tid];
    for (int c = 0; c < (1 + NREG) * DIM; ++c)
      acc += sh_concat[c] * Wo[c * HIDN + tid];
    out[(size_t)b * HIDN + tid] = fmaxf(acc, 0.f);
  }
}

extern "C" void kernel_launch(void* const* d_in, const int* in_sizes, int n_in,
                              void* d_out, int out_size, void* d_ws, size_t ws_size,
                              hipStream_t stream) {
  const float* influence = (const float*)d_in[0];
  const float* cstat     = (const float*)d_in[1];
  const float* Wc        = (const float*)d_in[2];
  const float* bc        = (const float*)d_in[3];
  const float* Wqkv      = (const float*)d_in[4];
  const float* bqkv      = (const float*)d_in[5];
  const float* Wa        = (const float*)d_in[6];
  const float* ba        = (const float*)d_in[7];
  const float* Wo        = (const float*)d_in[8];
  const float* bo        = (const float*)d_in[9];
  float* out = (float*)d_out;
  _Float16* ws = (_Float16*)d_ws;

  const int batch = in_sizes[0] / (2 * S_TOK);

  ca_prep_kernel<<<(DIM * QKVN + 255) / 256, 256, 0, stream>>>(Wqkv, Wa, ws);
  ca_main_kernel<<<batch, 128, 0, stream>>>(influence, cstat, Wc, bc, bqkv, ba,
                                            Wo, bo, ws, ws + QKVN * DIM, out);
}
